// SemanticCorrelationModule_89154931131017
// MI455X (gfx1250) — matrix operand order repaired and checked by reading, not compiled, not run on hardware
//
#include <hip/hip_runtime.h>
#include <hip/hip_bf16.h>

// SemanticCorrelationModule fused implementation for gfx1250 (MI455X).
// B=4, C=512, H=W=64 (N=4096), K=20 (pad 32), Cp=128.
// All GEMMs on v_wmma_f32_16x16x32_f16 (f16 in, f32 accumulate).
// Round 3:
//  - two-pass exact softmax: rowmax pass has zero cross-lane VALU in the loop;
//    accumulate pass uses fixed m (no rescale, per-lane partial row sums)
//  - split-K (x2) accumulate pass -> 2048 waves (~2/SIMD) for latency hiding
//  - w_proj pre-converted to f16 (kills 8 v_cvt per WMMA in the proj GEMM)
//  - block-cooperative double-buffered LDS tiles (row-major + transposed)

typedef _Float16 half_t;
typedef __attribute__((ext_vector_type(16))) _Float16 v16h;
typedef __attribute__((ext_vector_type(8)))  _Float16 v8h;
typedef __attribute__((ext_vector_type(8)))  float    v8f;

#define BSZ 4
#define CIN 512
#define NPIX 4096
#define KCLS 20
#define KPAD 32
#define CP 128
#define NSPLIT 2
#define KEYS_PER_SPLIT (NPIX / NSPLIT)

// LDS row strides (in halves), padded for bank spread + 16B alignment.
#define VROW_STRIDE 136   // 272B rows
#define VCOL_STRIDE 40    // 80B rows

__device__ __forceinline__ v8f wmma16(v16h a, v16h b, v8f c) {
  return __builtin_amdgcn_wmma_f32_16x16x32_f16(false, a, false, b, (short)0, c,
                                                false, false);
}

// A-fragment K index (16-bit A, 16x32): lanes 0-15 K {0..15};
// lanes 16-31 K {16..31}. Row M = lane&15.
__device__ __forceinline__ int kA(int e, int lhalf) {
  return lhalf * 16 + e;
}

__device__ __forceinline__ v8f vzero() {
  v8f z = {0.f, 0.f, 0.f, 0.f, 0.f, 0.f, 0.f, 0.f};
  return z;
}

// Build a v16h from two 16B-aligned 8-half chunks.
__device__ __forceinline__ v16h pack16(const half_t* lo, const half_t* hi) {
  const v8h a = *(const v8h*)lo;
  const v8h b = *(const v8h*)hi;
  v16h r;
#pragma unroll
  for (int e = 0; e < 8; ++e) { r[e] = a[e]; r[e + 8] = b[e]; }
  return r;
}

// ---------------------------------------------------------------------------
// Kernel 0: w_proj f32 -> f16 (one-time, 64K elements)
// ---------------------------------------------------------------------------
__global__ __launch_bounds__(256) void cvtw_kernel(const float* __restrict__ w,
                                                   half_t* __restrict__ wh) {
  const int i = blockIdx.x * 256 + threadIdx.x;
  wh[i] = (half_t)w[i];
}

// ---------------------------------------------------------------------------
// Kernel 1: P[b,n,:] = feat[b,:,n] . w_proj^T + b_proj   -> f16 Ph [B][N][Cp]
// One wave = 16 rows x 128 cols tile. grid 128 x block 256 (8 waves).
// ---------------------------------------------------------------------------
__global__ __launch_bounds__(256) void proj_kernel(
    const float* __restrict__ feat, const half_t* __restrict__ wh,
    const float* __restrict__ bias, half_t* __restrict__ Ph) {
  const int lane  = threadIdx.x & 31;
  const int wv    = threadIdx.x >> 5;
  const int lmod  = lane & 15;
  const int lhalf = lane >> 4;
  const int gw    = blockIdx.x * 8 + wv;
  const int b     = gw >> 8;
  const int n0    = (gw & 255) << 4;

  const float* featb = feat + (size_t)b * CIN * NPIX;

  v8f acc[8];
#pragma unroll
  for (int j = 0; j < 8; ++j) acc[j] = vzero();

  for (int c0 = 0; c0 < CIN; c0 += 32) {
    // A fragment: A[m][k] = feat[b][c0+k][n0+m]
    v16h a;
#pragma unroll
    for (int e = 0; e < 16; ++e) {
      const int k = kA(e, lhalf);
      a[e] = (half_t)featb[(size_t)(c0 + k) * NPIX + n0 + lmod];
    }
#pragma unroll
    for (int j = 0; j < 8; ++j) {
      // B fragment: B[k][o'] = wh[o0+o'][c0+k] -> contiguous 16 halves/lane
      const half_t* wr = wh + (size_t)(j * 16 + lmod) * CIN + c0 + lhalf * 16;
      acc[j] = wmma16(a, pack16(wr, wr + 8), acc[j]);
    }
  }
#pragma unroll
  for (int j = 0; j < 8; ++j) {
    const float bv = bias[j * 16 + lmod];
#pragma unroll
    for (int r = 0; r < 8; ++r) {
      const int m = r + lhalf * 8;
      Ph[((size_t)b * NPIX + n0 + m) * CP + j * 16 + lmod] =
          (half_t)(acc[j][r] + bv);
    }
  }
}

// ---------------------------------------------------------------------------
// Kernel 2: class_term[b,k,d] = sum_n cam[b,k,n] * P[b,n,d]  (K padded to 32)
// ---------------------------------------------------------------------------
__global__ __launch_bounds__(512) void classterm_kernel(
    const float* __restrict__ cam, const half_t* __restrict__ Ph,
    half_t* __restrict__ CTh) {
  const int lane  = threadIdx.x & 31;
  const int wv    = threadIdx.x >> 5;
  const int lmod  = lane & 15;
  const int lhalf = lane >> 4;
  const int b     = blockIdx.x;
  const int ktile = wv >> 3;
  const int dtile = wv & 7;

  const float*  camb = cam + (size_t)b * KCLS * NPIX;
  const half_t* Pb   = Ph + (size_t)b * NPIX * CP;

  v8f acc = vzero();
  const int kcls = ktile * 16 + lmod;
  const int d    = dtile * 16 + lmod;

  for (int n0 = 0; n0 < NPIX; n0 += 32) {
    v16h a;
#pragma unroll
    for (int e = 0; e < 16; ++e) {
      const int kn = kA(e, lhalf);
      a[e] = (kcls < KCLS) ? (half_t)camb[(size_t)kcls * NPIX + n0 + kn]
                           : (half_t)0.0f;
    }
    v16h bf;
#pragma unroll
    for (int e = 0; e < 16; ++e) {
      const int kn = e + lhalf * 16;  // B-frag K index
      bf[e] = Pb[(size_t)(n0 + kn) * CP + d];
    }
    acc = wmma16(a, bf, acc);
  }
#pragma unroll
  for (int r = 0; r < 8; ++r) {
    const int row = ktile * 16 + r + lhalf * 8;
    CTh[((size_t)b * KPAD + row) * CP + dtile * 16 + lmod] = (half_t)acc[r];
  }
}

// ---------------------------------------------------------------------------
// Kernel 3: rowmax[b][n] = max_m (P[b,n,:] . P[b,m,:])
// Loop has zero cross-lane VALU: per-lane running max, one reduce at the end.
// ---------------------------------------------------------------------------
__global__ __launch_bounds__(256) void rowmax_kernel(
    const half_t* __restrict__ Ph, float* __restrict__ rowmax) {
  __shared__ half_t vrow[2][32 * VROW_STRIDE];

  const int lane  = threadIdx.x & 31;
  const int wv    = threadIdx.x >> 5;
  const int lmod  = lane & 15;
  const int lhalf = lane >> 4;
  const int gw    = blockIdx.x * 8 + wv;
  const int b     = gw >> 8;
  const int n0    = (gw & 255) << 4;

  const half_t* Pb = Ph + (size_t)b * NPIX * CP;

  const int tr = threadIdx.x >> 3;   // 0..31
  const int ts = threadIdx.x & 7;    // 0..7
  auto stage_row = [&](int m0, int buf) {
    const v16h vv = *(const v16h*)(Pb + (size_t)(m0 + tr) * CP + ts * 16);
    *(v8h*)(&vrow[buf][tr * VROW_STRIDE + ts * 16])     = *(const v8h*)&vv;
    *(v8h*)(&vrow[buf][tr * VROW_STRIDE + ts * 16 + 8]) = ((const v8h*)&vv)[1];
  };

  v16h qa[4];
  {
    const half_t* qrow = Pb + (size_t)(n0 + lmod) * CP;
#pragma unroll
    for (int c = 0; c < 4; ++c)
      qa[c] = pack16(qrow + 32 * c + lhalf * 16, qrow + 32 * c + lhalf * 16 + 8);
  }

  float vmax[8];
#pragma unroll
  for (int r = 0; r < 8; ++r) vmax[r] = -1e30f;

  stage_row(0, 0);
  __syncthreads();

  for (int m0 = 0; m0 < NPIX; m0 += 32) {
    const int buf = (m0 >> 5) & 1;
    if (m0 + 32 < NPIX) stage_row(m0 + 32, buf ^ 1);

    v8f s0 = vzero(), s1 = vzero();
#pragma unroll
    for (int c = 0; c < 4; ++c) {
      const half_t* r0 = &vrow[buf][lmod * VROW_STRIDE + 32 * c + lhalf * 16];
      const half_t* r1 =
          &vrow[buf][(16 + lmod) * VROW_STRIDE + 32 * c + lhalf * 16];
      s0 = wmma16(qa[c], pack16(r0, r0 + 8), s0);
      s1 = wmma16(qa[c], pack16(r1, r1 + 8), s1);
    }
#pragma unroll
    for (int r = 0; r < 8; ++r)
      vmax[r] = fmaxf(vmax[r], fmaxf(s0[r], s1[r]));

    __syncthreads();
  }

#pragma unroll
  for (int r = 0; r < 8; ++r) {
    float v = vmax[r];
#pragma unroll
    for (int m = 1; m < 16; m <<= 1) v = fmaxf(v, __shfl_xor(v, m, 32));
    if (lmod == 0) rowmax[(size_t)b * NPIX + n0 + r + lhalf * 8] = v;
  }
}

// ---------------------------------------------------------------------------
// Kernel 4: split-K fixed-m flash accumulate.
//   Opart[y][b][n][d] = sum_{m in split y} exp(S[n][m]-rowmax[n]) * P[m][d]
//   Lpart[y][b][n]    = sum_{m in split y} exp(S[n][m]-rowmax[n])
// grid (128, NSPLIT) x 256. No cross-lane VALU and no rescale in the loop.
// ---------------------------------------------------------------------------
__global__ __launch_bounds__(256) void flashacc_kernel(
    const half_t* __restrict__ Ph, const float* __restrict__ rowmax,
    float* __restrict__ Opart, float* __restrict__ Lpart) {
  __shared__ half_t vrow[2][32 * VROW_STRIDE];
  __shared__ half_t vcol[2][CP * VCOL_STRIDE];
  __shared__ half_t stage[8][16 * 32];          // per-wave prob staging

  const int lane  = threadIdx.x & 31;
  const int wv    = threadIdx.x >> 5;
  const int lmod  = lane & 15;
  const int lhalf = lane >> 4;
  const int gw    = blockIdx.x * 8 + wv;
  const int b     = gw >> 8;
  const int n0    = (gw & 255) << 4;
  const int y     = blockIdx.y;
  const int ks    = y * KEYS_PER_SPLIT;

  const half_t* Pb  = Ph + (size_t)b * NPIX * CP;
  half_t*       stg = &stage[wv][0];

  const int tr = threadIdx.x >> 3;
  const int ts = threadIdx.x & 7;
  auto stage_tile = [&](int key0, int buf) {
    const v16h vv = *(const v16h*)(Pb + (size_t)(key0 + tr) * CP + ts * 16);
    *(v8h*)(&vrow[buf][tr * VROW_STRIDE + ts * 16])     = *(const v8h*)&vv;
    *(v8h*)(&vrow[buf][tr * VROW_STRIDE + ts * 16 + 8]) = ((const v8h*)&vv)[1];
#pragma unroll
    for (int e = 0; e < 16; ++e)
      vcol[buf][(ts * 16 + e) * VCOL_STRIDE + tr] = vv[e];
  };

  v16h qa[4];
  {
    const half_t* qrow = Pb + (size_t)(n0 + lmod) * CP;
#pragma unroll
    for (int c = 0; c < 4; ++c)
      qa[c] = pack16(qrow + 32 * c + lhalf * 16, qrow + 32 * c + lhalf * 16 + 8);
  }

  float mrow[8], lrow[8];
#pragma unroll
  for (int r = 0; r < 8; ++r) {
    mrow[r] = rowmax[(size_t)b * NPIX + n0 + r + lhalf * 8];
    lrow[r] = 0.f;
  }
  v8f o_acc[8];
#pragma unroll
  for (int j = 0; j < 8; ++j) o_acc[j] = vzero();

  stage_tile(ks, 0);
  __syncthreads();

  for (int mo = 0; mo < KEYS_PER_SPLIT; mo += 32) {
    const int buf = (mo >> 5) & 1;
    if (mo + 32 < KEYS_PER_SPLIT) stage_tile(ks + mo + 32, buf ^ 1);

    v8f s0 = vzero(), s1 = vzero();
#pragma unroll
    for (int c = 0; c < 4; ++c) {
      const half_t* r0 = &vrow[buf][lmod * VROW_STRIDE + 32 * c + lhalf * 16];
      const half_t* r1 =
          &vrow[buf][(16 + lmod) * VROW_STRIDE + 32 * c + lhalf * 16];
      s0 = wmma16(qa[c], pack16(r0, r0 + 8), s0);
      s1 = wmma16(qa[c], pack16(r1, r1 + 8), s1);
    }

    // Fixed-m exponentials; per-lane partial row sums (no shuffles here).
#pragma unroll
    for (int r = 0; r < 8; ++r) {
      const float p0 = __expf(s0[r] - mrow[r]);
      const float p1 = __expf(s1[r] - mrow[r]);
      lrow[r] += p0 + p1;
      s0[r] = p0; s1[r] = p1;
    }

    // C/D layout -> A-fragment layout via per-wave LDS slab (row stride 32).
#pragma unroll
    for (int r = 0; r < 8; ++r) {
      const int m = r + lhalf * 8;
      stg[m * 32 + lmod]      = (half_t)s0[r];
      stg[m * 32 + 16 + lmod] = (half_t)s1[r];
    }
    asm volatile("s_wait_dscnt 0x0" ::: "memory");
    const v16h pa = pack16(stg + lmod * 32 + lhalf * 16,
                           stg + lmod * 32 + lhalf * 16 + 8);
    asm volatile("s_wait_dscnt 0x0" ::: "memory");

    // O += prob . V  (contiguous reads from the transposed tile)
#pragma unroll
    for (int j = 0; j < 8; ++j) {
      const half_t* vc = &vcol[buf][(j * 16 + lmod) * VCOL_STRIDE + lhalf * 16];
      o_acc[j] = wmma16(pa, pack16(vc, vc + 8), o_acc[j]);
    }

    __syncthreads();
  }

  // Reduce per-lane row sums once; store partials.
#pragma unroll
  for (int r = 0; r < 8; ++r) {
    float rs = lrow[r];
#pragma unroll
    for (int m = 1; m < 16; m <<= 1) rs += __shfl_xor(rs, m, 32);
    if (lmod == 0)
      Lpart[((size_t)y * BSZ + b) * NPIX + n0 + r + lhalf * 8] = rs;
  }
#pragma unroll
  for (int j = 0; j < 8; ++j)
#pragma unroll
    for (int r = 0; r < 8; ++r)
      Opart[(((size_t)y * BSZ + b) * NPIX + n0 + r + lhalf * 8) * CP +
            j * 16 + lmod] = o_acc[j][r];
}

// ---------------------------------------------------------------------------
// Kernel 5: combine splits, normalize, project onto class_term, add cam.
//   out[b,k,n] = cam[b,k,n] + sum_d CT[b,k,d] * (sum_y O[y])/(sum_y L[y])
// ---------------------------------------------------------------------------
__global__ __launch_bounds__(256) void combine_kernel(
    const float* __restrict__ cam, const float* __restrict__ Opart,
    const float* __restrict__ Lpart, const half_t* __restrict__ CTh,
    float* __restrict__ out) {
  __shared__ half_t stage[8][16 * 128];

  const int lane  = threadIdx.x & 31;
  const int wv    = threadIdx.x >> 5;
  const int lmod  = lane & 15;
  const int lhalf = lane >> 4;
  const int gw    = blockIdx.x * 8 + wv;
  const int b     = gw >> 8;
  const int n0    = (gw & 255) << 4;

  half_t* stg = &stage[wv][0];

  float linv[8];
#pragma unroll
  for (int r = 0; r < 8; ++r) {
    const size_t rowi = (size_t)b * NPIX + n0 + r + lhalf * 8;
    linv[r] = 1.0f / (Lpart[rowi] + Lpart[(size_t)BSZ * NPIX + rowi]);
  }

  // Sum partials, normalize, stage as f16 pixel_term tile (row stride 128).
#pragma unroll
  for (int j = 0; j < 8; ++j) {
#pragma unroll
    for (int r = 0; r < 8; ++r) {
      const size_t rowi = (size_t)b * NPIX + n0 + r + lhalf * 8;
      const float o = Opart[rowi * CP + j * 16 + lmod] +
                      Opart[((size_t)BSZ * NPIX + rowi) * CP + j * 16 + lmod];
      stg[(r + lhalf * 8) * 128 + j * 16 + lmod] = (half_t)(o * linv[r]);
    }
  }
  asm volatile("s_wait_dscnt 0x0" ::: "memory");

  v16h ptf[4];
#pragma unroll
  for (int c = 0; c < 4; ++c)
    ptf[c] = pack16(stg + lmod * 128 + 32 * c + lhalf * 16,
                    stg + lmod * 128 + 32 * c + lhalf * 16 + 8);

  const half_t* CTb = CTh + (size_t)b * KPAD * CP;
  v8f oa[2] = {vzero(), vzero()};
#pragma unroll
  for (int t = 0; t < 2; ++t) {
#pragma unroll
    for (int c = 0; c < 4; ++c) {
      const half_t* ct =
          CTb + (size_t)(t * 16 + lmod) * CP + 32 * c + lhalf * 16;
      oa[t] = wmma16(ptf[c], pack16(ct, ct + 8), oa[t]);
    }
  }

#pragma unroll
  for (int t = 0; t < 2; ++t) {
    const int k = t * 16 + lmod;
    if (k < KCLS) {
#pragma unroll
      for (int r = 0; r < 8; ++r) {
        const size_t idx = ((size_t)b * KCLS + k) * NPIX + n0 + r + lhalf * 8;
        out[idx] = cam[idx] + oa[t][r];
      }
    }
  }
}

// ---------------------------------------------------------------------------
extern "C" void kernel_launch(void* const* d_in, const int* in_sizes, int n_in,
                              void* d_out, int out_size, void* d_ws,
                              size_t ws_size, hipStream_t stream) {
  const float* feat = (const float*)d_in[0];  // [4,512,64,64]
  const float* cam  = (const float*)d_in[1];  // [4,20,64,64]
  const float* wpr  = (const float*)d_in[2];  // [128,512]
  const float* bpr  = (const float*)d_in[3];  // [128]
  float* out = (float*)d_out;                 // [4,20,64,64]

  char* w8 = (char*)d_ws;
  half_t* Ph    = (half_t*)(w8);                               // 4 MiB
  half_t* CTh   = (half_t*)(w8 + (4u << 20));                  // 32 KiB
  half_t* wh    = (half_t*)(w8 + (4u << 20) + (64u << 10));    // 128 KiB
  float*  rowmx = (float*) (w8 + (4u << 20) + (256u << 10));   // 64 KiB
  float*  Lpart = (float*) (w8 + (4u << 20) + (384u << 10));   // 128 KiB
  float*  Opart = (float*) (w8 + (5u << 20));                  // 16 MiB

  cvtw_kernel<<<CP * CIN / 256, 256, 0, stream>>>(wpr, wh);
  proj_kernel<<<128, 256, 0, stream>>>(feat, wh, bpr, Ph);
  classterm_kernel<<<BSZ, 512, 0, stream>>>(cam, Ph, CTh);
  rowmax_kernel<<<128, 256, 0, stream>>>(Ph, rowmx);
  flashacc_kernel<<<dim3(128, NSPLIT), 256, 0, stream>>>(Ph, rowmx, Opart,
                                                         Lpart);
  combine_kernel<<<128, 256, 0, stream>>>(cam, Opart, Lpart, CTh, out);
}
